// MultiHeadAttention_33457795236556
// MI455X (gfx1250) — matrix operand order repaired and checked by reading, not compiled, not run on hardware
//
#include <hip/hip_runtime.h>
#include <hip/hip_bf16.h>
#include <math.h>

// ---------------------------------------------------------------------------
// Multi-head attention on MI455X (gfx1250, wave32, WMMA + TDM).
//   B=4, S=4 -> BS=16 slices, L=1024, D=1024, H=16, depth=64
// Pipeline:
//   1) gemm_kernel<f32 A, headsplit> x3 : Q/K/V proj. LDS-tiled bf16 WMMA,
//      double-buffered; weight tile prefetched by Tensor Data Mover
//      (tensor_load_to_lds, TENSORcnt) one k-chunk ahead of the WMMAs.
//   2) attn_kernel : per (bs, 16-row q tile), loop heads: logits (WMMA) ->
//      softmax (LDS + shfl) -> attn_mean accumulate -> ctx = attn@V
//      (8 waves: 4 depth tiles x 2-way K-split, LDS reduction).
//   3) gemm_kernel<bf16 A, flat f32 out> : out = ctx@wo + bo.
// ---------------------------------------------------------------------------

#define BSZ     16      // B*S
#define LSEQ    1024
#define DMODEL  1024
#define NHEAD   16
#define HDEPTH  64
#define NEGBIG  (-1.0e9f)

typedef __attribute__((ext_vector_type(16))) __bf16        v16bf;
typedef __attribute__((ext_vector_type(8)))  __bf16        v8bf;
typedef __attribute__((ext_vector_type(8)))  float         v8f;
typedef __attribute__((ext_vector_type(4)))  unsigned int  u32x4;
typedef __attribute__((ext_vector_type(8)))  int           i32x8;
typedef __attribute__((ext_vector_type(4)))  int           i32x4;
typedef __bf16 bf16_t;

// ---- WMMA fragment index helpers (ISA 7.12.2, wave32, 16-bit operands) ----
// A (16x32 MxK): lane&15 = M, half=lane>>4 selects K-range;
//   VGPR v<4: K = 2v + 8*half; v>=4: K = 2v+8 + 8*half (pairs +0/+1).
// B (32x16 KxN): lane&15 = N; same K mapping as A (VGPR v: K = a_frag_k(v,half)).
// C/D (16x16 f32): VGPR r -> row r + 8*half, col = lane&15.
__device__ __forceinline__ int a_frag_k(int v, int half) {
    return 2 * v + (v >= 4 ? 8 : 0) + 8 * half;
}
__device__ __forceinline__ int b_frag_k(int v, int half) {
    return 2 * v + (v >= 4 ? 8 : 0) + 8 * half;
}

// ---------------------------------------------------------------------------
// Tensor Data Mover: async DMA of a 2D f32 tile (rows x cols, row stride in
// elements) from global memory into LDS. D# layout per CDNA5 ISA.
// ---------------------------------------------------------------------------
#if defined(__has_builtin)
#  if __has_builtin(__builtin_amdgcn_tensor_load_to_lds) && \
      __has_builtin(__builtin_amdgcn_s_wait_tensorcnt)
#    define USE_TDM 1
#  endif
#endif
#ifndef USE_TDM
#  define USE_TDM 0
#endif

#if USE_TDM
__device__ __forceinline__ void tdm_load_2d_f32(unsigned lds_byte_addr,
                                                const void* gptr,
                                                unsigned cols, unsigned rows,
                                                unsigned long long stride_elems)
{
    const unsigned long long ga = (unsigned long long)(uintptr_t)gptr;
    u32x4 g0;
    g0[0] = 1u;                                   // count=1, user mode
    g0[1] = lds_byte_addr;                        // lds_addr
    g0[2] = (unsigned)(ga & 0xffffffffu);         // global_addr[31:0]
    g0[3] = (unsigned)((ga >> 32) & 0x01ffffffu)  // global_addr[56:32]
          | (2u << 30);                           // type = 2 ("image")
    i32x8 g1;
    g1[0] = (int)(2u << 16);                      // data_size = 2 -> 4 bytes
    g1[1] = (int)((cols & 0xffffu) << 16);        // tensor_dim0[15:0]
    g1[2] = (int)(((cols >> 16) & 0xffffu)        // tensor_dim0[31:16]
          | ((rows & 0xffffu) << 16));            // tensor_dim1[15:0]
    g1[3] = (int)(((rows >> 16) & 0xffffu)        // tensor_dim1[31:16]
          | ((cols & 0xffffu) << 16));            // tile_dim0 = cols
    g1[4] = (int)(rows & 0xffffu);                // tile_dim1 = rows, tile_dim2=0
    g1[5] = (int)(stride_elems & 0xffffffffu);    // tensor_dim0_stride[31:0]
    g1[6] = (int)((stride_elems >> 32) & 0xffffu);// tensor_dim0_stride[47:32]
    g1[7] = 0;
    const i32x4 gz = {0, 0, 0, 0};
#if defined(__clang_major__) && (__clang_major__ >= 23)
    const i32x8 gz8 = {0, 0, 0, 0, 0, 0, 0, 0};
    __builtin_amdgcn_tensor_load_to_lds(g0, g1, gz, gz, gz8, 0);
#else
    __builtin_amdgcn_tensor_load_to_lds(g0, g1, gz, gz, 0);
#endif
}
#endif // USE_TDM

// ===========================================================================
// LDS-tiled GEMM: D[16384,1024] = A[16384,1024] @ W[1024,1024] + bias
//   Block tile 64(M) x 128(N), k-chunk 32; 8 waves as 2(M) x 4(N),
//   each wave owns a 2x2 grid of 16x16 WMMA tiles (4 WMMAs / k-chunk).
//   Double-buffered LDS: chunk i+1 is staged (TDM for W, coalesced b128
//   loads for A) while chunk i feeds the WMMAs; one barrier per iteration.
// ===========================================================================
template <bool A_IS_F32, bool HEADSPLIT>
__global__ __launch_bounds__(256)
void gemm_kernel(const void* __restrict__ Aptr, const float* __restrict__ W,
                 const float* __restrict__ bias, void* __restrict__ Dptr)
{
    __shared__ __align__(16) __bf16 Asmem[2][64 * 32];    // 2 x 4 KB
    __shared__ __align__(16) float  Bsmem[2][32 * 128];   // 2 x 16 KB

    const int tid   = threadIdx.x;
    const int lane  = tid & 31;
    const int wave  = tid >> 5;
    const int half  = lane >> 4;
    const int lm    = lane & 15;
    const int waveM = wave >> 2;             // 0..1
    const int waveN = wave & 3;              // 0..3
    const int rowBase = blockIdx.y * 64;
    const int colBase = blockIdx.x * 128;

    // A-staging assignment: 8 contiguous elements per thread
    const int ar = tid >> 2;                 // 0..63
    const int ac = (tid & 3) * 8;            // 0,8,16,24

    // ---- stage one k-chunk into LDS buffer `buf` --------------------------
    auto stage = [&](int kc, int buf) {
#if USE_TDM
        if (wave == 0) {
            tdm_load_2d_f32((unsigned)(uintptr_t)&Bsmem[buf][0],
                            W + (size_t)kc * DMODEL + colBase,
                            128u, 32u, (unsigned long long)DMODEL);
        }
#else
        {
            const int br = tid >> 3;               // 0..31
            const int bc = (tid & 7) * 16;         // 16 floats per thread
            const float* gw = W + (size_t)(kc + br) * DMODEL + colBase + bc;
            float* lw = &Bsmem[buf][br * 128 + bc];
#pragma unroll
            for (int i = 0; i < 4; ++i)
                *(float4*)(lw + 4 * i) = *(const float4*)(gw + 4 * i);
        }
#endif
        v8bf av;
        if (A_IS_F32) {
            const float* ga =
                (const float*)Aptr + (size_t)(rowBase + ar) * DMODEL + kc + ac;
            const float4 f0 = *(const float4*)ga;
            const float4 f1 = *(const float4*)(ga + 4);
            av[0] = (__bf16)f0.x; av[1] = (__bf16)f0.y;
            av[2] = (__bf16)f0.z; av[3] = (__bf16)f0.w;
            av[4] = (__bf16)f1.x; av[5] = (__bf16)f1.y;
            av[6] = (__bf16)f1.z; av[7] = (__bf16)f1.w;
        } else {
            const bf16_t* ga =
                (const bf16_t*)Aptr + (size_t)(rowBase + ar) * DMODEL + kc + ac;
            av = *(const v8bf*)ga;
        }
        *(v8bf*)&Asmem[buf][ar * 32 + ac] = av;
    };

    v8f acc[2][2] = {};

    stage(0, 0);
#if USE_TDM
    if (wave == 0) __builtin_amdgcn_s_wait_tensorcnt(0);
#endif
    __syncthreads();

    const int NCHUNK = DMODEL / 32;          // 32
    for (int i = 0; i < NCHUNK; ++i) {
        const int cur = i & 1;
        // ---- prefetch next chunk into the other buffer --------------------
        if (i + 1 < NCHUNK) stage((i + 1) * 32, cur ^ 1);

        // ---- build fragments from LDS buffer `cur`, 4 WMMAs ---------------
        v16bf afr[2], bfr[2];
#pragma unroll
        for (int mt = 0; mt < 2; ++mt) {
            const int arow = waveM * 32 + mt * 16 + lm;
#pragma unroll
            for (int v = 0; v < 8; ++v) {
                const int ka = a_frag_k(v, half);
                afr[mt][2 * v]     = Asmem[cur][arow * 32 + ka];
                afr[mt][2 * v + 1] = Asmem[cur][arow * 32 + ka + 1];
            }
        }
#pragma unroll
        for (int nt = 0; nt < 2; ++nt) {
            const int bcol = waveN * 32 + nt * 16 + lm;
#pragma unroll
            for (int v = 0; v < 8; ++v) {
                const int kb = b_frag_k(v, half);
                bfr[nt][2 * v]     = (__bf16)Bsmem[cur][kb * 128 + bcol];
                bfr[nt][2 * v + 1] = (__bf16)Bsmem[cur][(kb + 1) * 128 + bcol];
            }
        }
#pragma unroll
        for (int mt = 0; mt < 2; ++mt)
#pragma unroll
            for (int nt = 0; nt < 2; ++nt)
                acc[mt][nt] = __builtin_amdgcn_wmma_f32_16x16x32_bf16(
                    false, afr[mt], false, bfr[nt], (short)0, acc[mt][nt],
                    false, false);

#if USE_TDM
        if (wave == 0 && i + 1 < NCHUNK)
            __builtin_amdgcn_s_wait_tensorcnt(0);
#endif
        __syncthreads();   // next-buffer staging complete / safe to overwrite
    }

    // ---- epilogue: bias + store -------------------------------------------
#pragma unroll
    for (int nt = 0; nt < 2; ++nt) {
        const int col  = colBase + waveN * 32 + nt * 16 + lm;
        const float bv = bias[col];
#pragma unroll
        for (int mt = 0; mt < 2; ++mt) {
#pragma unroll
            for (int r = 0; r < 8; ++r) {
                const int row = rowBase + waveM * 32 + mt * 16 + r + 8 * half;
                const float val = acc[mt][nt][r] + bv;
                if (HEADSPLIT) {
                    const int bs = row >> 10, lpos = row & 1023;
                    const int h = col >> 6, dd = col & 63;
                    ((bf16_t*)Dptr)[(((size_t)(bs * NHEAD + h) * LSEQ) + lpos)
                                        * HDEPTH + dd] = (bf16_t)val;
                } else {
                    ((float*)Dptr)[(size_t)row * DMODEL + col] = val;
                }
            }
        }
    }
}

// ===========================================================================
// Attention: block = (q-tile of 16 rows, bs); loops over all 16 heads.
// Dynamic LDS: logits[16][1024] f32 (64 KB) + ctxbuf[16][64] f32 (4 KB).
// ===========================================================================
__global__ __launch_bounds__(256)
void attn_kernel(const bf16_t* __restrict__ Qh, const bf16_t* __restrict__ Kh,
                 const bf16_t* __restrict__ Vh, const float* __restrict__ mask,
                 bf16_t* __restrict__ ctx, float* __restrict__ attn_mean)
{
    extern __shared__ float smemf[];
    float* logits = smemf;                 // 16 * 1024
    float* ctxbuf = smemf + 16 * LSEQ;     // 16 * 64 (K-split partials)

    const int tid  = threadIdx.x;
    const int lane = tid & 31;
    const int wave = tid >> 5;
    const int half = lane >> 4;
    const int lm   = lane & 15;
    const int bs   = blockIdx.y;           // 0..15
    const int q0   = blockIdx.x * 16;      // query-row tile base
    const float scale = 0.125f;            // 1/sqrt(64)

    for (int h = 0; h < NHEAD; ++h) {
        const bf16_t* Qb = Qh + ((size_t)(bs * NHEAD + h) * LSEQ) * HDEPTH;
        const bf16_t* Kb = Kh + ((size_t)(bs * NHEAD + h) * LSEQ) * HDEPTH;
        const bf16_t* Vb = Vh + ((size_t)(bs * NHEAD + h) * LSEQ) * HDEPTH;

        // ---- Q fragments: constant across all key tiles of this head ------
        v16bf qfr[2];
#pragma unroll
        for (int kk = 0; kk < 2; ++kk) {
#pragma unroll
            for (int v = 0; v < 8; ++v) {
                const int ka = a_frag_k(v, half);
                const bf16_t* ap =
                    Qb + (size_t)(q0 + lm) * HDEPTH + kk * 32 + ka;
                qfr[kk][2 * v]     = ap[0];
                qfr[kk][2 * v + 1] = ap[1];
            }
        }

        // ---- logits tile [16,1024]: 64 key-tiles over 8 waves -------------
        for (int ktile = wave; ktile < LSEQ / 16; ktile += 8) {
            v8f c = {};
#pragma unroll
            for (int kk = 0; kk < 2; ++kk) {
                v16bf b;
#pragma unroll
                for (int v = 0; v < 8; ++v) {
                    const int kb = b_frag_k(v, half);
                    // B[k][n] = K[key=n][k]: contiguous depth pair per lane
                    const bf16_t* bp =
                        Kb + (size_t)(ktile * 16 + lm) * HDEPTH + kk * 32 + kb;
                    b[2 * v]     = bp[0];
                    b[2 * v + 1] = bp[1];
                }
                c = __builtin_amdgcn_wmma_f32_16x16x32_bf16(
                        false, qfr[kk], false, b, (short)0, c, false, false);
            }
            const int key = ktile * 16 + lm;
#pragma unroll
            for (int r = 0; r < 8; ++r) {
                const int row = r + 8 * half;
                const float mv =
                    mask[((size_t)bs * LSEQ + (q0 + row)) * LSEQ + key];
                logits[row * LSEQ + key] = c[r] * scale + mv * NEGBIG;
            }
        }
        __syncthreads();

        // ---- softmax: 16 rows, 16 lanes/row (shfl_xor in 16-lane group) ---
        {
            const int row = tid >> 4;      // 2 rows per wave
            const int seg = tid & 15;      // 64 elements per thread
            float* lr = logits + row * LSEQ + seg * 64;
            float mx = -3.0e38f;
#pragma unroll 8
            for (int i = 0; i < 64; ++i) mx = fmaxf(mx, lr[i]);
#pragma unroll
            for (int off = 8; off >= 1; off >>= 1)
                mx = fmaxf(mx, __shfl_xor(mx, off, 32));
            float s = 0.0f;
#pragma unroll 8
            for (int i = 0; i < 64; ++i) {
                const float e = __expf(lr[i] - mx);
                lr[i] = e;
                s += e;
            }
#pragma unroll
            for (int off = 8; off >= 1; off >>= 1)
                s += __shfl_xor(s, off, 32);
            const float inv = 1.0f / s;
            float* am = attn_mean +
                        ((size_t)bs * LSEQ + (q0 + row)) * LSEQ + seg * 64;
#pragma unroll 8
            for (int i = 0; i < 64; ++i) {
                const float p = lr[i] * inv;
                lr[i] = p;
                const float acc = (h == 0) ? 0.0f : am[i];
                am[i] = acc + p * (1.0f / (float)NHEAD);
            }
        }
        __syncthreads();

        // ---- ctx = attn @ V : 8 waves = 4 depth tiles x 2-way K-split -----
        {
            const int ntile = wave & 3;
            const int khalf = wave >> 2;
            v8f c = {};
            const int k0 = khalf * (LSEQ / 2);
            for (int kc = k0; kc < k0 + LSEQ / 2; kc += 32) {
                v16bf a, b;
#pragma unroll
                for (int v = 0; v < 8; ++v) {
                    const int ka = a_frag_k(v, half);
                    const float* ap = logits + lm * LSEQ + kc + ka;
                    a[2 * v]     = (__bf16)ap[0];
                    a[2 * v + 1] = (__bf16)ap[1];
                    const int kb = b_frag_k(v, half);
                    // B[k][n] = V[key=k][depth n]: pair strided by HDEPTH
                    const bf16_t* bp =
                        Vb + (size_t)(kc + kb) * HDEPTH + ntile * 16 + lm;
                    b[2 * v]     = bp[0];
                    b[2 * v + 1] = bp[HDEPTH];
                }
                c = __builtin_amdgcn_wmma_f32_16x16x32_bf16(
                        false, a, false, b, (short)0, c, false, false);
            }
            // reduce the two K-halves via LDS, then store bf16 ctx
            if (khalf == 1) {
#pragma unroll
                for (int r = 0; r < 8; ++r)
                    ctxbuf[(r + 8 * half) * HDEPTH + ntile * 16 + lm] = c[r];
            }
            __syncthreads();
            if (khalf == 0) {
#pragma unroll
                for (int r = 0; r < 8; ++r) {
                    const int row = r + 8 * half;
                    const int col = h * HDEPTH + ntile * 16 + lm;
                    const float val =
                        c[r] + ctxbuf[row * HDEPTH + ntile * 16 + lm];
                    ctx[((size_t)bs * LSEQ + (q0 + row)) * DMODEL + col] =
                        (bf16_t)val;
                }
            }
        }
        __syncthreads();   // logits/ctxbuf LDS reused by next head
    }
}

// ===========================================================================
extern "C" void kernel_launch(void* const* d_in, const int* in_sizes, int n_in,
                              void* d_out, int out_size, void* d_ws,
                              size_t ws_size, hipStream_t stream)
{
    const float* q    = (const float*)d_in[0];
    const float* k    = (const float*)d_in[1];
    const float* v    = (const float*)d_in[2];
    const float* mask = (const float*)d_in[3];
    const float* wq   = (const float*)d_in[4];
    const float* bq   = (const float*)d_in[5];
    const float* wk   = (const float*)d_in[6];
    const float* bk   = (const float*)d_in[7];
    const float* wv   = (const float*)d_in[8];
    const float* bv   = (const float*)d_in[9];
    const float* wo   = (const float*)d_in[10];
    const float* bo   = (const float*)d_in[11];

    float* out       = (float*)d_out;                         // [16,1024,1024]
    float* attn_mean = out + (size_t)BSZ * LSEQ * DMODEL;     // [16,1024,1024]

    // workspace: Qh | Kh | Vh | ctx, each bf16 [16,16,1024,64] = 33.5 MB
    const size_t perBuf = (size_t)BSZ * NHEAD * LSEQ * HDEPTH;
    bf16_t* Qh  = (bf16_t*)d_ws;
    bf16_t* Kh  = Qh + perBuf;
    bf16_t* Vh  = Kh + perBuf;
    bf16_t* ctx = Vh + perBuf;

    const dim3 blk(256);
    const dim3 grd(DMODEL / 128, (BSZ * LSEQ) / 64);   // 8 x 256 blocks

    gemm_kernel<true, true><<<grd, blk, 0, stream>>>(q, wq, bq, Qh);
    gemm_kernel<true, true><<<grd, blk, 0, stream>>>(k, wk, bk, Kh);
    gemm_kernel<true, true><<<grd, blk, 0, stream>>>(v, wv, bv, Vh);

    const size_t smem = (size_t)(16 * LSEQ + 16 * HDEPTH) * sizeof(float);
    attn_kernel<<<dim3(LSEQ / 16, BSZ), 256, smem, stream>>>(
        Qh, Kh, Vh, mask, ctx, attn_mean);

    gemm_kernel<false, false><<<grd, blk, 0, stream>>>(ctx, wo, bo, out);
}
